// CoucheinitialeGNN_5497558139184
// MI455X (gfx1250) — matrix operand order repaired and checked by reading, not compiled, not run on hardware
//
#include <hip/hip_runtime.h>
#include <hip/hip_bf16.h>

#define N_NODES 100000
#define N_EDGES 1600000
#define NTILES  (N_EDGES / 16)

typedef __attribute__((ext_vector_type(16))) _Float16 v16h;
typedef __attribute__((ext_vector_type(8)))  _Float16 v8h;
typedef __attribute__((ext_vector_type(8)))  float    v8f;
typedef __attribute__((ext_vector_type(4)))  float    v4f;

// ---------------------------------------------------------------- zero d accumulator
__global__ void gnn_zero_kernel(float* __restrict__ p, int n4) {
    int i = blockIdx.x * blockDim.x + threadIdx.x;
    if (i < n4) ((v4f*)p)[i] = (v4f){0.f, 0.f, 0.f, 0.f};
}

// ---------------------------------------------------------------- prep: W2^T padded to 64x128 f16, b2 padded to 64
__global__ void gnn_prep_kernel(const float* __restrict__ W2, const float* __restrict__ b2,
                                _Float16* __restrict__ w2t, float* __restrict__ b2p) {
    int g = blockIdx.x * blockDim.x + threadIdx.x;
    if (g < 64 * 128) {
        int n = g >> 7;      // output column (0..63), pad >=54
        int k = g & 127;     // K index
        w2t[g] = (n < 54) ? (_Float16)W2[k * 54 + n] : (_Float16)0.f;
    }
    if (g < 64) b2p[g] = (g < 54) ? b2[g] : 0.f;
}

// ---------------------------------------------------------------- main: WMMA tile kernel
// NORM=false: accumulate segment-sum d via atomics. NORM=true: recompute w, divide by d[src], store.
template <bool NORM>
__global__ __launch_bounds__(128) void gnn_pass_kernel(
    const float* __restrict__ dist, const long long* __restrict__ srcidx,
    const _Float16* __restrict__ w2t, const float* __restrict__ b2p,
    const float* __restrict__ W1, const float* __restrict__ b1,
    float* __restrict__ d, float* __restrict__ out) {
    // W2^T staged per block: 64 cols x K=128 halves, row stride 136 halves (272B) -> bank-spread
    __shared__ alignas(16) _Float16 w2tsh[64 * 136];
    // per-wave f32 tile for coalesced writeback (NORM pass), 16 rows x stride 68 floats
    __shared__ alignas(16) float wtsh[4][16 * 68];
    __shared__ float dsh[4][16];
    __shared__ int   ssh[4][16];

    const int  tid  = threadIdx.x;
    const int  w    = tid >> 5;
    const int  lane = tid & 31;
    const int  M    = lane & 15;
    const bool lo   = lane < 16;
    const int  off8 = lo ? 0 : 8;

    // cooperative copy of W2^T into padded LDS (1024 x 16B chunks / 128 threads)
    for (int c = tid; c < 1024; c += 128) {
        int n = c >> 4, kc = (c & 15) * 8;
        *(v8h*)&w2tsh[n * 136 + kc] = *(const v8h*)(w2t + n * 128 + kc);
    }
    __syncthreads();

    // Per-lane W1/b1 as packed f16 for direct in-register A-fragment generation.
    // A 16x32 f16 layout: lane holds M=lane&15; this lane's K-set for step s is
    //   {s*32+off8+0..7} U {s*32+16+off8+0..7}  (off8 = 0 for lanes<16, 8 otherwise)
    v16h w1h[4], b1h[4];
#pragma unroll
    for (int s = 0; s < 4; ++s) {
        const int k0 = s * 32 + off8;
        const int k1 = k0 + 16;
        v16h wv, bv;
#pragma unroll
        for (int i = 0; i < 8; ++i) {
            wv[i]     = (_Float16)W1[k0 + i];
            wv[8 + i] = (_Float16)W1[k1 + i];
            bv[i]     = (_Float16)b1[k0 + i];
            bv[8 + i] = (_Float16)b1[k1 + i];
        }
        w1h[s] = wv;
        b1h[s] = bv;
    }
    float b2v[4];
#pragma unroll
    for (int t = 0; t < 4; ++t) b2v[t] = b2p[t * 16 + M];

    for (int tile = blockIdx.x * 4 + w; tile < NTILES; tile += gridDim.x * 4) {
        // stop LICM from hoisting the loop-invariant LDS B loads back into 128 registers
        asm volatile("" ::: "memory");
        const int tb = tile * 16;
        if (lo) {
            dsh[w][M] = dist[tb + M];
            ssh[w][M] = (int)srcidx[tb + M];
        }
        const float    de  = dsh[w][M];   // dist of edge M (lane generates A rows for edge M)
        const _Float16 deh = (_Float16)de;

        // --- all four A fragments up front (distinct registers -> no WMMA WAR hazard nops) ---
        v16h Afrag[4];
#pragma unroll
        for (int s = 0; s < 4; ++s) {
            v16h A;
#pragma unroll
            for (int i = 0; i < 16; ++i) {
                _Float16 x = (_Float16)(deh * w1h[s][i] + b1h[s][i]);
                A[i] = (x > (_Float16)0) ? x : (_Float16)0;
            }
            Afrag[s] = A;
        }

        // --- WMMA: acc = relu(dist*W1+b1) @ W2p, starting from inline-zero C ---
        v8f C[4];
#pragma unroll
        for (int t = 0; t < 4; ++t) C[t] = (v8f)(0.f);   // folds to WMMA src2 = inline 0
#pragma unroll
        for (int s = 0; s < 4; ++s) {
#pragma unroll
            for (int t = 0; t < 4; ++t) {
                // B 32x16 f16 layout: lanes 0-15 hold K=0..7,16..23 of col N=lane; lanes 16-31 K=8..15,24..31
                const _Float16* bp = &w2tsh[(t * 16 + M) * 136 + s * 32 + (lo ? 0 : 8)];
                v8h blo = *(const v8h*)bp;
                v8h bhi = *(const v8h*)(bp + 16);
                v16h Bst;
#pragma unroll
                for (int i = 0; i < 8; ++i) { Bst[i] = blo[i]; Bst[8 + i] = bhi[i]; }
                C[t] = __builtin_amdgcn_wmma_f32_16x16x32_f16(false, Afrag[s], false, Bst,
                                                              (short)0, C[t], false, false);
            }
        }
        // fold bias in after the K loop: mlp_out = relu(acc + b2)
#pragma unroll
        for (int t = 0; t < 4; ++t)
#pragma unroll
            for (int r = 0; r < 8; ++r) C[t][r] = fmaxf(C[t][r] + b2v[t], 0.f);

        if (!NORM) {
            // one-hot contribution: +1.0 at d[src][bucket]
            if (lo) {
                int b = min((int)de, 9);
                atomicAdd(&d[(size_t)ssh[w][M] * 64 + b], 1.0f);
            }
            // mlp contribution: C layout: row = r + (lo?0:8), col = 10 + t*16 + M
#pragma unroll
            for (int r = 0; r < 8; ++r) {
                const int row = r + off8;
                float* drow = &d[(size_t)ssh[w][row] * 64 + 10];
#pragma unroll
                for (int t = 0; t < 4; ++t) {
                    if (t < 3 || M < 6) {  // mlp col t*16+M < 54
                        float v = C[t][r];
                        if (v > 0.f) atomicAdd(drow + t * 16 + M, v);
                    }
                }
            }
        } else {
            // transpose the tile through LDS for coalesced output
            float* wt = &wtsh[w][0];  // 16 rows x stride 68 floats
#pragma unroll
            for (int r = 0; r < 8; ++r) {
                const int row = r + off8;
#pragma unroll
                for (int t = 0; t < 4; ++t) wt[row * 68 + t * 16 + M] = C[t][r];
            }
            for (int i = 0; i < 32; ++i) {
                const int row = i >> 1;
                const int col = (i & 1) * 32 + lane;  // 0..63, coalesced 128B per half-row
                const int s   = ssh[w][row];
                float wv;
                if (col < 10) {
                    int b = min((int)dsh[w][row], 9);
                    wv = (col == b) ? 1.0f : 0.0f;
                } else {
                    wv = wt[row * 68 + (col - 10)];
                }
                const float dv = d[(size_t)s * 64 + col];  // L2-resident gather (regular temporal)
                const float o  = (dv != 0.f) ? (wv / dv) : 0.f;
                // 410MB write-once stream: non-temporal so it doesn't evict d from L2
                __builtin_nontemporal_store(o, &out[(size_t)(tb + row) * 64 + col]);
            }
        }
    }
}

// ---------------------------------------------------------------- launch
extern "C" void kernel_launch(void* const* d_in, const int* in_sizes, int n_in,
                              void* d_out, int out_size, void* d_ws, size_t ws_size,
                              hipStream_t stream) {
    // inputs: 0:x(unused) 1:edge_index(i64,2xE) 2:edge_attr(f32,E) 3:W1(128) 4:b1(128) 5:W2(128x54) 6:b2(54)
    const long long* ei   = (const long long*)d_in[1];  // row 0 = src
    const float*     dist = (const float*)d_in[2];
    const float*     W1   = (const float*)d_in[3];
    const float*     b1   = (const float*)d_in[4];
    const float*     W2   = (const float*)d_in[5];
    const float*     b2   = (const float*)d_in[6];
    float*           out  = (float*)d_out;

    // workspace layout
    float*    dacc = (float*)d_ws;                                   // 100000*64 f32 = 25.6 MB
    _Float16* w2t  = (_Float16*)((char*)d_ws + (size_t)N_NODES * 64 * 4);      // 64*128 f16
    float*    b2p  = (float*)((char*)d_ws + (size_t)N_NODES * 64 * 4 + 64 * 128 * 2);

    gnn_zero_kernel<<<(N_NODES * 64 / 4 + 255) / 256, 256, 0, stream>>>(dacc, N_NODES * 64 / 4);
    gnn_prep_kernel<<<32, 256, 0, stream>>>(W2, b2, w2t, b2p);
    gnn_pass_kernel<false><<<4096, 128, 0, stream>>>(dist, ei, w2t, b2p, W1, b1, dacc, out);
    gnn_pass_kernel<true ><<<4096, 128, 0, stream>>>(dist, ei, w2t, b2p, W1, b1, dacc, out);
}
